// GraphAttentionConv_26087631356322
// MI455X (gfx1250) — matrix operand order repaired and checked by reading, not compiled, not run on hardware
//
#include <hip/hip_runtime.h>
#include <hip/hip_bf16.h>
#include <math.h>

typedef __attribute__((ext_vector_type(16))) _Float16 v16h;
typedef __attribute__((ext_vector_type(8)))  _Float16 v8h;
typedef __attribute__((ext_vector_type(8)))  float    v8f;
typedef __attribute__((ext_vector_type(2)))  float    v2f;

#define N_NODES 8192
#define FIN     256
#define FOUT    128
#define NEG_SLOPE 0.01f

// LDS adj tile: 16 rows x 32 ints, rows padded to 36 dwords (144B) so the
// per-lane ds_load_b128 of 16 distinct rows spreads across banks.
#define TROW 36

// ---------------------------------------------------------------------------
// Phase A1: Xp = X @ W^T + b  via V_WMMA_F32_16X16X4_F32 (keeps f32 precision)
// Also emits XpT in f16, column-major ([FOUT][N]) so the main GEMM's
// B-fragments are contiguous 32B per lane.  One wave per 16x16 output tile.
// ---------------------------------------------------------------------------
__global__ __launch_bounds__(32)
void xp_gemm_kernel(const float* __restrict__ X, const float* __restrict__ W,
                    const float* __restrict__ bias,
                    float* __restrict__ Xp, _Float16* __restrict__ XpT) {
  const int lane = threadIdx.x;
  const int r0 = blockIdx.x * 16;        // 512 M-tiles
  const int n0 = blockIdx.y * 16;        // 8 N-tiles
  const int half = lane >> 4;            // 0 | 1
  const int l15  = lane & 15;

  // A 16x4 f32: lane = M, VGPR0/1 = K {0,1} (lanes 0-15) / {2,3} (16-31)
  // B 4x16  f32: lane = N, VGPR0/1 = K {0,1} (lanes 0-15) / {2,3} (16-31)
  const float* xrow = X + (size_t)(r0 + l15) * FIN;
  const float* wrow = W + (size_t)(n0 + l15) * FIN;   // W is (FOUT, FIN) row-major

  v8f c = {};
  for (int k = 0; k < FIN; k += 4) {
    const int kp = k + half * 2;
    v2f a = { xrow[kp], xrow[kp + 1] };
    v2f b = { wrow[kp], wrow[kp + 1] };
    c = __builtin_amdgcn_wmma_f32_16x16x4_f32(false, a, false, b, (short)0, c,
                                              false, false);
  }

  const float bv = bias[n0 + l15];
  const int rbase = r0 + half * 8;       // C: VGPR r -> M = r (+8 for hi lanes)
  v8h hx;
#pragma unroll
  for (int r = 0; r < 8; ++r) {
    const float v = c[r] + bv;
    Xp[(size_t)(rbase + r) * FOUT + (n0 + l15)] = v;
    hx[r] = (_Float16)v;
  }
  *(v8h*)(XpT + (size_t)(n0 + l15) * N_NODES + rbase) = hx;
}

// ---------------------------------------------------------------------------
// Phase A2: as = Xp @ S[0:128], ad = Xp @ S[128:256]
// ---------------------------------------------------------------------------
__global__ __launch_bounds__(256)
void attn_coef_kernel(const float* __restrict__ Xp, const float* __restrict__ S,
                      float* __restrict__ as_, float* __restrict__ ad_) {
  const int i = blockIdx.x * blockDim.x + threadIdx.x;
  if (i >= N_NODES) return;
  const float* row = Xp + (size_t)i * FOUT;
  float s1 = 0.f, s2 = 0.f;
#pragma unroll 4
  for (int f = 0; f < FOUT; ++f) {
    const float v = row[f];
    s1 += v * S[f];
    s2 += v * S[FOUT + f];
  }
  as_[i] = s1;
  ad_[i] = s2;
}

// ---------------------------------------------------------------------------
// Phase A3: admax = max_j ad[j] (global softmax shift; diagonal is always
// unmasked so denominators stay finite)
// ---------------------------------------------------------------------------
__global__ __launch_bounds__(256)
void admax_kernel(const float* __restrict__ ad_, float* __restrict__ admax) {
  __shared__ float red[256];
  float m = -INFINITY;
  for (int i = threadIdx.x; i < N_NODES; i += 256) m = fmaxf(m, ad_[i]);
  red[threadIdx.x] = m;
  __syncthreads();
  for (int s = 128; s > 0; s >>= 1) {
    if (threadIdx.x < s) red[threadIdx.x] = fmaxf(red[threadIdx.x], red[threadIdx.x + s]);
    __syncthreads();
  }
  if (threadIdx.x == 0) admax[0] = red[0];
}

// ---------------------------------------------------------------------------
// Phase B (HBM-bound on adj): adj tiles stream into LDS with double-buffered
// global_load_async_to_lds_b128 (ASYNCcnt), probs -> f16 WMMA accumulate of
// attn @ Xp with per-row sums, sigmoid(acc / rowsum) epilogue.
// Block = 256 threads (8 waves), one 16-row tile per block; each wave owns a
// 1024-column strip in K=32 chunks; partials merged via ds_add_f32 in LDS.
// ---------------------------------------------------------------------------
__global__ __launch_bounds__(256)
void gat_fused_kernel(const int* __restrict__ adj, const _Float16* __restrict__ XpT,
                      const float* __restrict__ as_, const float* __restrict__ ad_,
                      const float* __restrict__ admax, float* __restrict__ out) {
  __shared__ __align__(16) int adjtile[8][2][16 * TROW];  // 36.9 KB
  __shared__ float acc[16][FOUT];                         // 8 KB
  __shared__ float rowsum[16];
  const int tid  = threadIdx.x;
  const int wave = tid >> 5;
  const int lane = tid & 31;
  const int r0   = blockIdx.x * 16;

  for (int idx = tid; idx < 16 * FOUT; idx += 256) (&acc[0][0])[idx] = 0.f;
  if (tid < 16) rowsum[tid] = 0.f;
  __syncthreads();

  const int m     = lane & 15;            // row owned by this lane
  const int hb    = (lane >> 4) * 16;     // A-frag K base within chunk: 0 | 16
  const int row_g = r0 + m;
  const float a_s = as_[row_g];
  const float t0  = a_s + admax[0];
  const float shift = fmaxf(t0, NEG_SLOPE * t0);

  // per-lane async-copy pattern: 4 instructions x 32 lanes = 128 x 16B segs
  // seg layout: u = t*32+lane, row = u>>3, seg16 = u&7; LDS row stride 144B.
  const int arow = (lane >> 3);           // +t*4 per instruction
  const int aseg = (lane & 7);
  const int* gbase = adj + (size_t)r0 * N_NODES;

  auto issue_async = [&](int bb, int j) {
#pragma unroll
    for (int t = 0; t < 4; ++t) {
      const int row = t * 4 + arow;
      const unsigned lds = (unsigned)(size_t)&adjtile[wave][bb][0]
                         + (unsigned)(row * (TROW * 4) + aseg * 16);
      const int* g = gbase + (size_t)row * N_NODES + j + aseg * 4;
      asm volatile("global_load_async_to_lds_b128 %0, %1, off"
                   :: "v"(lds), "v"(g) : "memory");
    }
  };

  v8f c[8] = {};
  float lsum = 0.f;

  const int jstart = wave * (N_NODES / 8);
  const int jend   = jstart + (N_NODES / 8);

  issue_async(0, jstart);
  int pb = 0;
  for (int j = jstart; j < jend; j += 32) {
    if (j + 32 < jend) {
      issue_async(pb ^ 1, j + 32);
      asm volatile("s_wait_asynccnt 0x4" ::: "memory");   // current tile ready
    } else {
      asm volatile("s_wait_asynccnt 0x0" ::: "memory");
    }

    const int* tp = &adjtile[wave][pb][m * TROW + hb];
    const int4 a0 = *(const int4*)(tp);
    const int4 a1 = *(const int4*)(tp + 4);
    const int4 a2 = *(const int4*)(tp + 8);
    const int4 a3 = *(const int4*)(tp + 12);
    const float4* adp = (const float4*)(ad_ + j + hb);
    const float4 d0 = adp[0], d1 = adp[1], d2 = adp[2], d3 = adp[3];

    const int col0 = j + hb;
    float ls = lsum;
    auto pcalc = [&](int a, float adv, int col) -> _Float16 {
      const bool keep = (a != 0) | (col == row_g);
      const float t = a_s + adv;
      const float e = fmaxf(t, NEG_SLOPE * t);            // leaky_relu
      const float pv = keep ? __expf(e - shift) : 0.f;
      ls += pv;
      return (_Float16)pv;
    };

    // v16h element 2r+h <-> VGPR r half h. A k order chosen here (contiguous, not AMD's table order):
    // lanes 0-15: K = 0..15; lanes 16-31: K = 16..31 (matches B).
    v16h af;
    af[0]  = pcalc(a0.x, d0.x, col0 + 0);
    af[1]  = pcalc(a0.y, d0.y, col0 + 1);
    af[2]  = pcalc(a0.z, d0.z, col0 + 2);
    af[3]  = pcalc(a0.w, d0.w, col0 + 3);
    af[4]  = pcalc(a1.x, d1.x, col0 + 4);
    af[5]  = pcalc(a1.y, d1.y, col0 + 5);
    af[6]  = pcalc(a1.z, d1.z, col0 + 6);
    af[7]  = pcalc(a1.w, d1.w, col0 + 7);
    af[8]  = pcalc(a2.x, d2.x, col0 + 8);
    af[9]  = pcalc(a2.y, d2.y, col0 + 9);
    af[10] = pcalc(a2.z, d2.z, col0 + 10);
    af[11] = pcalc(a2.w, d2.w, col0 + 11);
    af[12] = pcalc(a3.x, d3.x, col0 + 12);
    af[13] = pcalc(a3.y, d3.y, col0 + 13);
    af[14] = pcalc(a3.z, d3.z, col0 + 14);
    af[15] = pcalc(a3.w, d3.w, col0 + 15);
    lsum = ls;

    // B k order chosen here (same as A): lane = N column; lanes 0-15 hold K 0..15,
    // lanes 16-31 hold K 16..31 -> one contiguous 32B load from XpT (L2-hot).
    const _Float16* bbase = XpT + (size_t)(lane & 15) * N_NODES
                                + j + (lane >> 4) * 16;
#pragma unroll
    for (int nt = 0; nt < 8; ++nt) {
      const v16h bf = *(const v16h*)(bbase + (size_t)nt * 16 * N_NODES);
      c[nt] = __builtin_amdgcn_wmma_f32_16x16x32_f16(false, af, false, bf,
                                                     (short)0, c[nt], false, false);
    }
    pb ^= 1;
  }

  // merge partials across the 8 waves (ds_add_f32 atomics)
  atomicAdd(&rowsum[m], lsum);
  const int rb = (lane >> 4) * 8;        // C: VGPR r -> M = r (+8 hi lanes)
#pragma unroll
  for (int nt = 0; nt < 8; ++nt)
#pragma unroll
    for (int r = 0; r < 8; ++r)
      atomicAdd(&acc[rb + r][nt * 16 + (lane & 15)], c[nt][r]);
  __syncthreads();

  for (int idx = tid; idx < 16 * FOUT; idx += 256) {
    const int rr = idx >> 7, cc = idx & (FOUT - 1);
    const float v = acc[rr][cc] / rowsum[rr];
    out[(size_t)(r0 + rr) * FOUT + cc] = 1.f / (1.f + __expf(-v));
  }
}

// ---------------------------------------------------------------------------
extern "C" void kernel_launch(void* const* d_in, const int* in_sizes, int n_in,
                              void* d_out, int out_size, void* d_ws, size_t ws_size,
                              hipStream_t stream) {
  (void)in_sizes; (void)n_in; (void)out_size; (void)ws_size;
  const float* X   = (const float*)d_in[0];   // (8192, 256) f32
  const int*   adj = (const int*)d_in[1];     // (8192, 8192) i32
  const float* W   = (const float*)d_in[2];   // (128, 256) f32
  const float* b   = (const float*)d_in[3];   // (128,) f32
  const float* S   = (const float*)d_in[4];   // (256,) f32
  float* out = (float*)d_out;                 // (8192, 128) f32

  // workspace layout (~6.1 MB total)
  char* ws = (char*)d_ws;
  _Float16* XpT = (_Float16*)ws;                                // 2 MB
  float* Xp    = (float*)(ws + (size_t)2 * 1024 * 1024);        // 4 MB
  float* as_   = (float*)(ws + (size_t)6 * 1024 * 1024);        // 32 KB
  float* ad_   = as_ + N_NODES;                                 // 32 KB
  float* admax = ad_ + N_NODES;                                 // 4 B

  xp_gemm_kernel<<<dim3(N_NODES / 16, FOUT / 16), dim3(32), 0, stream>>>(X, W, b, Xp, XpT);
  attn_coef_kernel<<<N_NODES / 256, 256, 0, stream>>>(Xp, S, as_, ad_);
  admax_kernel<<<1, 256, 0, stream>>>(ad_, admax);
  gat_fused_kernel<<<N_NODES / 16, 256, 0, stream>>>(adj, XpT, as_, ad_, admax, out);
}
